// TransformerOp_60988535603895
// MI455X (gfx1250) — matrix operand order repaired and checked by reading, not compiled, not run on hardware
//
#include <hip/hip_runtime.h>
#include <hip/hip_bf16.h>
#include <math.h>

// ---------------------------------------------------------------------------
// Transformer block (linear attention path, L=4096) for gfx1250 (MI455X).
// Big GEMMs (~412 of ~420 GFLOP) run on v_wmma_f32_16x16x32_bf16 with f32
// accumulation; fp32 operands are converted with native bf16 cvt on the LDS
// staging path (float2 loads -> packed bf16 stores).
// ---------------------------------------------------------------------------

#define Bsz 4
#define Lsz 4096
#define Dsz 1024
#define Hn  8
#define HD  128
#define ROWS (Bsz * Lsz)          // 16384
#define LDS_STRIDE 34             // 32 + 2 pad (keeps dword alignment)

typedef __attribute__((ext_vector_type(16))) __bf16       v16bf;
typedef __attribute__((ext_vector_type(8)))  float        v8f;
typedef __attribute__((ext_vector_type(8)))  unsigned int v8u;

union FragU { v8u u; v16bf b; };

// Native conversions: clang lowers float->__bf16 fptrunc to the gfx1250
// V_CVT_*BF16* instructions (RNE), replacing the manual integer rounding.
__device__ __forceinline__ unsigned short bf16_bits(float f) {
  union { __bf16 b; unsigned short u; } c; c.b = (__bf16)f; return c.u;
}
__device__ __forceinline__ unsigned int pack_bf16x2(float lo, float hi) {
  union { __bf16 b[2]; unsigned int u; } c;
  c.b[0] = (__bf16)lo; c.b[1] = (__bf16)hi; return c.u;
}

// A-matrix fragment, 16x32 bf16 (ISA 7.12.2):
// lane<16: row=lane,    K = {0..7, 16..23};  lane>=16: row=lane-16, K = {8..15, 24..31}
__device__ __forceinline__ v16bf lds_frag_a(const unsigned short* base, int row0, int lane) {
  const int r  = row0 + (lane & 15);
  const int kb = (lane & 16) >> 1;              // 0 or 8
  const unsigned short* p = base + r * LDS_STRIDE + kb;
  FragU f;
  f.u[0] = *(const unsigned int*)(p + 0);
  f.u[1] = *(const unsigned int*)(p + 2);
  f.u[2] = *(const unsigned int*)(p + 4);
  f.u[3] = *(const unsigned int*)(p + 6);
  f.u[4] = *(const unsigned int*)(p + 16);
  f.u[5] = *(const unsigned int*)(p + 18);
  f.u[6] = *(const unsigned int*)(p + 20);
  f.u[7] = *(const unsigned int*)(p + 22);
  return f.b;
}

// B-matrix fragment, 32x16 bf16 (same K order as A):
// lane<16: col=lane, K={0..7,16..23}; lane>=16: col=lane-16, K={8..15,24..31}. LDS holds B transposed [n][k].
__device__ __forceinline__ v16bf lds_frag_b(const unsigned short* base, int col0, int lane) {
  const int n  = col0 + (lane & 15);
  const int kb = (lane & 16) >> 1;              // 0 or 8
  const unsigned short* p = base + n * LDS_STRIDE + kb;
  FragU f;
#pragma unroll
  for (int v = 0; v < 8; ++v) f.u[v] = *(const unsigned int*)(p + 2 * v + (v >= 4 ? 8 : 0));
  return f.b;
}

__device__ __forceinline__ v8f wmma_bf16(v16bf a, v16bf b, v8f c) {
  return __builtin_amdgcn_wmma_f32_16x16x32_bf16(false, a, false, b, (short)0, c, false, false);
}

// ---------------------------------------------------------------------------
// Generic GEMM: C[MxN] = A[MxK] * B[KxN], all fp32 row-major in global memory.
// Block tile 128x128, BK=32, 256 threads = 8 waves in a 2(M) x 4(N) grid;
// each wave owns 4x2 WMMA tiles. EPI: 0 = none, 1 = exact GELU.
// Grid: x = N/128, y = M/128. Dims must be multiples of tile sizes (they are).
// ---------------------------------------------------------------------------
template <int EPI>
__global__ __launch_bounds__(256) void gemm_bf16_kernel(
    const float* __restrict__ A, const float* __restrict__ B, float* __restrict__ C,
    int K, int lda, int ldb, int ldc)
{
  __shared__ unsigned short As[128 * LDS_STRIDE];   // [m][k]
  __shared__ unsigned short Bs[128 * LDS_STRIDE];   // [n][k] (transposed)
  const int tid  = threadIdx.x;
  const int lane = tid & 31;
  const int wid  = tid >> 5;
  const int wM   = (wid >> 2) * 64;                 // 0 or 64
  const int wN   = (wid & 3) * 32;                  // 0,32,64,96
  const int m0   = blockIdx.y * 128;
  const int n0   = blockIdx.x * 128;

  v8f acc[4][2];
#pragma unroll
  for (int mt = 0; mt < 4; ++mt)
#pragma unroll
    for (int nt = 0; nt < 2; ++nt)
#pragma unroll
      for (int e = 0; e < 8; ++e) acc[mt][nt][e] = 0.0f;

  for (int k0 = 0; k0 < K; k0 += 32) {
    // A tile: 128x32, k-contiguous in global and LDS -> float2 + packed store
#pragma unroll
    for (int i = 0; i < 8; ++i) {
      int pidx = tid + 256 * i;                     // pair index (2048 pairs)
      int m = pidx >> 4, kk = (pidx & 15) * 2;
      float2 a2 = *(const float2*)&A[(size_t)(m0 + m) * lda + (k0 + kk)];
      *(unsigned int*)&As[m * LDS_STRIDE + kk] = pack_bf16x2(a2.x, a2.y);
    }
    // B tile: 32x128, n-contiguous in global, transposed [n][k] in LDS
#pragma unroll
    for (int i = 0; i < 8; ++i) {
      int pidx = tid + 256 * i;
      int kk = pidx >> 6, n = (pidx & 63) * 2;
      float2 b2 = *(const float2*)&B[(size_t)(k0 + kk) * ldb + (n0 + n)];
      Bs[n * LDS_STRIDE + kk]       = bf16_bits(b2.x);
      Bs[(n + 1) * LDS_STRIDE + kk] = bf16_bits(b2.y);
    }
    if (k0 + 32 < K) {                              // prefetch next k-tile
      __builtin_prefetch(&A[(size_t)(m0 + (tid >> 1)) * lda + (k0 + 32)], 0, 0);
      __builtin_prefetch(&B[(size_t)(k0 + 32 + (tid >> 3)) * ldb + n0 + (tid & 7) * 16], 0, 0);
    }
    __syncthreads();
    v16bf af[4], bf[2];
#pragma unroll
    for (int mt = 0; mt < 4; ++mt) af[mt] = lds_frag_a(As, wM + mt * 16, lane);
#pragma unroll
    for (int nt = 0; nt < 2; ++nt) bf[nt] = lds_frag_b(Bs, wN + nt * 16, lane);
#pragma unroll
    for (int mt = 0; mt < 4; ++mt)
#pragma unroll
      for (int nt = 0; nt < 2; ++nt)
        acc[mt][nt] = wmma_bf16(af[mt], bf[nt], acc[mt][nt]);
    __syncthreads();
  }

  const int nlocal = lane & 15;
  const int mofs   = (lane >> 4) * 8;               // C/D: VGPR v -> M = v + 8*half
#pragma unroll
  for (int mt = 0; mt < 4; ++mt)
#pragma unroll
    for (int nt = 0; nt < 2; ++nt)
#pragma unroll
      for (int v = 0; v < 8; ++v) {
        int row = m0 + wM + mt * 16 + mofs + v;
        int col = n0 + wN + nt * 16 + nlocal;
        float x = acc[mt][nt][v];
        if (EPI == 1) x = 0.5f * x * (1.0f + erff(x * 0.70710678118654752f));
        C[(size_t)row * ldc + col] = x;
      }
}

// ---------------------------------------------------------------------------
// elu(x)+1 feature map applied in place to the q and k slices of qkv.
// qkv row-major [ROWS, 3072]; q = cols [0,1024), k = [1024,2048).
// ---------------------------------------------------------------------------
__global__ __launch_bounds__(256) void featuremap_kernel(float* __restrict__ qkv) {
  size_t i   = (size_t)blockIdx.x * 256 + threadIdx.x;   // over ROWS*2048
  size_t row = i >> 11;
  int    c   = (int)(i & 2047);
  float* p = qkv + row * 3072 + c;
  float  x = *p;
  *p = (x > 0.0f) ? (x + 1.0f) : __expf(x);              // elu(x)+1
}

__global__ void zero_kernel(float* __restrict__ p, int n) {
  int i = blockIdx.x * 256 + threadIdx.x;
  if (i < n) p[i] = 0.0f;
}

// k_sum[bh][d] = sum_l k[b,l,h,d].  grid = (32 bh, 16 l-chunks), 256 threads.
__global__ __launch_bounds__(256) void ksum_kernel(const float* __restrict__ qkv,
                                                   float* __restrict__ ksum) {
  int bh = blockIdx.x, chunk = blockIdx.y;
  int b = bh >> 3, h = bh & 7;
  int d  = threadIdx.x & 127;
  int lo = threadIdx.x >> 7;
  const float* kp = qkv + (size_t)b * Lsz * 3072 + 1024 + h * 128 + d;
  float s = 0.0f;
  for (int i = 0; i < 128; ++i) {
    int l = chunk * 256 + lo * 128 + i;
    s += kp[(size_t)l * 3072];
  }
  atomicAdd(&ksum[bh * 128 + d], s);
}

// ---------------------------------------------------------------------------
// kv[bh][d][m] = sum_l k[l][d] * v[l][m].  One block per (b,h); K = L = 4096.
// k,v tiles transposed while staging into LDS.
// ---------------------------------------------------------------------------
__global__ __launch_bounds__(256) void kv_kernel(const float* __restrict__ qkv,
                                                 float* __restrict__ kvb) {
  __shared__ unsigned short As[128 * LDS_STRIDE];   // [d][l]
  __shared__ unsigned short Bs[128 * LDS_STRIDE];   // [m][l]
  const int tid = threadIdx.x, lane = tid & 31, wid = tid >> 5;
  const int wM = (wid >> 2) * 64, wN = (wid & 3) * 32;
  const int bh = blockIdx.x, b = bh >> 3, h = bh & 7;
  const float* kp = qkv + (size_t)b * Lsz * 3072 + 1024 + h * 128;
  const float* vp = qkv + (size_t)b * Lsz * 3072 + 2048 + h * 128;

  v8f acc[4][2];
#pragma unroll
  for (int mt = 0; mt < 4; ++mt)
#pragma unroll
    for (int nt = 0; nt < 2; ++nt)
#pragma unroll
      for (int e = 0; e < 8; ++e) acc[mt][nt][e] = 0.0f;

  for (int l0 = 0; l0 < Lsz; l0 += 32) {
#pragma unroll
    for (int i = 0; i < 8; ++i) {                   // 32(l) x 128(d) pairs along d
      int pidx = tid + 256 * i;
      int l = pidx >> 6, d = (pidx & 63) * 2;
      float2 k2 = *(const float2*)&kp[(size_t)(l0 + l) * 3072 + d];
      float2 v2 = *(const float2*)&vp[(size_t)(l0 + l) * 3072 + d];
      As[d * LDS_STRIDE + l]       = bf16_bits(k2.x);
      As[(d + 1) * LDS_STRIDE + l] = bf16_bits(k2.y);
      Bs[d * LDS_STRIDE + l]       = bf16_bits(v2.x);
      Bs[(d + 1) * LDS_STRIDE + l] = bf16_bits(v2.y);
    }
    __syncthreads();
    v16bf af[4], bf[2];
#pragma unroll
    for (int mt = 0; mt < 4; ++mt) af[mt] = lds_frag_a(As, wM + mt * 16, lane);
#pragma unroll
    for (int nt = 0; nt < 2; ++nt) bf[nt] = lds_frag_b(Bs, wN + nt * 16, lane);
#pragma unroll
    for (int mt = 0; mt < 4; ++mt)
#pragma unroll
      for (int nt = 0; nt < 2; ++nt)
        acc[mt][nt] = wmma_bf16(af[mt], bf[nt], acc[mt][nt]);
    __syncthreads();
  }
  const int nlocal = lane & 15, mofs = (lane >> 4) * 8;
#pragma unroll
  for (int mt = 0; mt < 4; ++mt)
#pragma unroll
    for (int nt = 0; nt < 2; ++nt)
#pragma unroll
      for (int v = 0; v < 8; ++v) {
        int d = wM + mt * 16 + mofs + v;
        int m = wN + nt * 16 + nlocal;
        kvb[(size_t)bh * 16384 + d * 128 + m] = acc[mt][nt][v];
      }
}

// ---------------------------------------------------------------------------
// out[l][m] = (q[l,:] @ kv[:,m]) / (q[l,:] @ ksum + 1e-8), heads re-merged.
// Grid: (L/128, B*H).  K = HD = 128 (4 WMMA k-steps).
// ---------------------------------------------------------------------------
__global__ __launch_bounds__(256) void attn_out_kernel(const float* __restrict__ qkv,
                                                       const float* __restrict__ kvb,
                                                       const float* __restrict__ ksum,
                                                       float* __restrict__ attn) {
  __shared__ unsigned short As[128 * LDS_STRIDE];   // q [l][d]
  __shared__ unsigned short Bs[128 * LDS_STRIDE];   // kv [m][d]
  __shared__ float pnorm[256];
  __shared__ float nrm[128];
  const int tid = threadIdx.x, lane = tid & 31, wid = tid >> 5;
  const int wM = (wid >> 2) * 64, wN = (wid & 3) * 32;
  const int bh = blockIdx.y, b = bh >> 3, h = bh & 7;
  const int l0 = blockIdx.x * 128;
  const float* qp  = qkv + (size_t)(b * Lsz + l0) * 3072 + h * 128;
  const float* kvp = kvb + (size_t)bh * 16384;
  const float* ks  = ksum + bh * 128;

  {                                                 // normalizer per row
    int r = tid >> 1, hf = tid & 1;
    float s = 0.0f;
    for (int d = hf * 64; d < hf * 64 + 64; ++d) s += qp[(size_t)r * 3072 + d] * ks[d];
    pnorm[tid] = s;
  }
  __syncthreads();
  if (tid < 128) nrm[tid] = pnorm[2 * tid] + pnorm[2 * tid + 1] + 1e-8f;
  __syncthreads();

  v8f acc[4][2];
#pragma unroll
  for (int mt = 0; mt < 4; ++mt)
#pragma unroll
    for (int nt = 0; nt < 2; ++nt)
#pragma unroll
      for (int e = 0; e < 8; ++e) acc[mt][nt][e] = 0.0f;

  for (int k0 = 0; k0 < 128; k0 += 32) {
#pragma unroll
    for (int i = 0; i < 8; ++i) {                   // q tile 128x32
      int pidx = tid + 256 * i;
      int m = pidx >> 4, dd = (pidx & 15) * 2;
      float2 q2 = *(const float2*)&qp[(size_t)m * 3072 + k0 + dd];
      *(unsigned int*)&As[m * LDS_STRIDE + dd] = pack_bf16x2(q2.x, q2.y);
    }
#pragma unroll
    for (int i = 0; i < 8; ++i) {                   // kv tile 32x128 -> [m][d]
      int pidx = tid + 256 * i;
      int kk = pidx >> 6, n = (pidx & 63) * 2;
      float2 b2 = *(const float2*)&kvp[(size_t)(k0 + kk) * 128 + n];
      Bs[n * LDS_STRIDE + kk]       = bf16_bits(b2.x);
      Bs[(n + 1) * LDS_STRIDE + kk] = bf16_bits(b2.y);
    }
    __syncthreads();
    v16bf af[4], bf[2];
#pragma unroll
    for (int mt = 0; mt < 4; ++mt) af[mt] = lds_frag_a(As, wM + mt * 16, lane);
#pragma unroll
    for (int nt = 0; nt < 2; ++nt) bf[nt] = lds_frag_b(Bs, wN + nt * 16, lane);
#pragma unroll
    for (int mt = 0; mt < 4; ++mt)
#pragma unroll
      for (int nt = 0; nt < 2; ++nt)
        acc[mt][nt] = wmma_bf16(af[mt], bf[nt], acc[mt][nt]);
    __syncthreads();
  }
  const int nlocal = lane & 15, mofs = (lane >> 4) * 8;
#pragma unroll
  for (int mt = 0; mt < 4; ++mt)
#pragma unroll
    for (int nt = 0; nt < 2; ++nt)
#pragma unroll
      for (int v = 0; v < 8; ++v) {
        int ml = wM + mt * 16 + mofs + v;
        int nl = wN + nt * 16 + nlocal;
        float x = acc[mt][nt][v] / nrm[ml];
        attn[(size_t)(b * Lsz + l0 + ml) * Dsz + h * 128 + nl] = x;
      }
}

// ---------------------------------------------------------------------------
// out = rms_norm(a + b, g) over rows of 1024. One block per row.
// ---------------------------------------------------------------------------
__global__ __launch_bounds__(256) void rmsnorm_kernel(const float* __restrict__ a,
                                                      const float* __restrict__ b,
                                                      const float* __restrict__ g,
                                                      float* __restrict__ out) {
  __shared__ float red[256];
  const int row = blockIdx.x, t = threadIdx.x;
  const float* pa = a + (size_t)row * Dsz;
  const float* pb = b + (size_t)row * Dsz;
  float v[4]; float s = 0.0f;
#pragma unroll
  for (int i = 0; i < 4; ++i) {
    int c = t + 256 * i;
    v[i] = pa[c] + pb[c];
    s += v[i] * v[i];
  }
  red[t] = s;
  __syncthreads();
  for (int off = 128; off > 0; off >>= 1) {
    if (t < off) red[t] += red[t + off];
    __syncthreads();
  }
  float scale = rsqrtf(red[0] * (1.0f / 1024.0f) + 1.1920929e-07f);
#pragma unroll
  for (int i = 0; i < 4; ++i) {
    int c = t + 256 * i;
    out[(size_t)row * Dsz + c] = v[i] * scale * g[c];
  }
}

// ---------------------------------------------------------------------------
extern "C" void kernel_launch(void* const* d_in, const int* in_sizes, int n_in,
                              void* d_out, int out_size, void* d_ws, size_t ws_size,
                              hipStream_t stream) {
  const float* x    = (const float*)d_in[0];
  const float* Wqkv = (const float*)d_in[1];
  const float* Wout = (const float*)d_in[2];
  const float* g1   = (const float*)d_in[3];
  const float* Wff1 = (const float*)d_in[4];
  const float* Wff2 = (const float*)d_in[5];
  const float* g2   = (const float*)d_in[6];
  float* out = (float*)d_out;
  float* ws  = (float*)d_ws;

  const size_t R = ROWS;                    // 16384
  float* regA = ws;                         // qkv [R,3072], later ff1 [R,4096]
  float* bufB = ws + R * 4096;              // attn_out, later h1     [R,1024]
  float* bufC = bufB + R * 1024;            // attn_proj, later ffn   [R,1024]
  float* kvb  = bufC + R * 1024;            // [32,128,128]
  float* ksum = kvb + 32 * 128 * 128;       // [32,128]

  dim3 blk(256);

  // 1) qkv = x @ W_qkv   [16384,1024] x [1024,3072]
  gemm_bf16_kernel<0><<<dim3(3072 / 128, ROWS / 128), blk, 0, stream>>>(
      x, Wqkv, regA, 1024, 1024, 3072, 3072);

  // 2) q,k <- elu(.)+1 in place
  featuremap_kernel<<<dim3((ROWS * 2048) / 256), blk, 0, stream>>>(regA);

  // 3) k_sum
  zero_kernel<<<dim3(16), blk, 0, stream>>>(ksum, 32 * 128);
  ksum_kernel<<<dim3(32, 16), blk, 0, stream>>>(regA, ksum);

  // 4) kv = k^T @ v per (b,h)
  kv_kernel<<<dim3(32), blk, 0, stream>>>(regA, kvb);

  // 5) attn = (q @ kv) / (q @ k_sum + 1e-8), heads merged -> bufB
  attn_out_kernel<<<dim3(Lsz / 128, 32), blk, 0, stream>>>(regA, kvb, ksum, bufB);

  // 6) attn_proj = attn @ W_out -> bufC
  gemm_bf16_kernel<0><<<dim3(1024 / 128, ROWS / 128), blk, 0, stream>>>(
      bufB, Wout, bufC, 1024, 1024, 1024, 1024);

  // 7) h1 = rms_norm(attn_proj + x, g1) -> bufB
  rmsnorm_kernel<<<dim3(ROWS), blk, 0, stream>>>(bufC, x, g1, bufB);

  // 8) ff1 = gelu(h1 @ W_ff1) -> regA
  gemm_bf16_kernel<1><<<dim3(4096 / 128, ROWS / 128), blk, 0, stream>>>(
      bufB, Wff1, regA, 1024, 1024, 4096, 4096);

  // 9) ffn = ff1 @ W_ff2 -> bufC
  gemm_bf16_kernel<0><<<dim3(1024 / 128, ROWS / 128), blk, 0, stream>>>(
      regA, Wff2, bufC, 4096, 4096, 1024, 1024);

  // 10) out = rms_norm(ffn + h1, g2)
  rmsnorm_kernel<<<dim3(ROWS), blk, 0, stream>>>(bufC, bufB, g2, out);
}
